// InformerClassifier_5196910428229
// MI455X (gfx1250) — matrix operand order repaired and checked by reading, not compiled, not run on hardware
//
#include <hip/hip_runtime.h>
#include <math.h>

// ---- problem constants (match reference) ----
#define B_    64
#define T_    2048
#define F_    32
#define DM    128
#define NH    4
#define HD    32
#define DFF   256
#define NCLS  16
#define UTOP  40          // min(5*ceil(ln 2048), 2048) = 40
#define ELAY  2

#define BM    128         // C-tile rows per block (8 waves x 16)
#define BN    64          // C-tile cols per block (4 WMMA subtiles)

typedef __bf16 bf16_t;
typedef __attribute__((ext_vector_type(16))) __bf16 bf16x16;
typedef __attribute__((ext_vector_type(8)))  __bf16 bf16x8;
typedef __attribute__((ext_vector_type(8)))  float  f32x8;
typedef __attribute__((ext_vector_type(4)))  int    v4i;

#define AS1 __attribute__((address_space(1)))
#define AS3 __attribute__((address_space(3)))

__device__ __forceinline__ bf16_t f2bf(float f) {
  union { float f; unsigned u; } v; v.f = f;
  unsigned r = v.u + 0x7FFFu + ((v.u >> 16) & 1u);   // round-to-nearest-even
  unsigned short s = (unsigned short)(r >> 16);
  union { unsigned short s; bf16_t b; } o; o.s = s; return o.b;
}

__device__ __forceinline__ float wave_sum32(float v) {
  #pragma unroll
  for (int m = 16; m > 0; m >>= 1) v += __shfl_xor(v, m);
  return v;
}

// ---- gfx1250 async global->LDS copy (ASYNCcnt-tracked), 16B per lane ----
__device__ __forceinline__ void async_copy_b128(const bf16_t* g, bf16_t* l) {
#if __has_builtin(__builtin_amdgcn_global_load_async_to_lds_b128)
  __builtin_amdgcn_global_load_async_to_lds_b128(
      (AS1 v4i*)(v4i*)(void*)const_cast<bf16_t*>(g),
      (AS3 v4i*)(v4i*)(void*)l, 0, 0);
#else
  unsigned lofs = (unsigned)(size_t)(AS3 char*)l;
  asm volatile("global_load_async_to_lds_b128 %0, %1, off"
               :: "v"(lofs), "v"(g) : "memory");
#endif
}

__device__ __forceinline__ void wait_async0() {
#if __has_builtin(__builtin_amdgcn_s_wait_asynccnt)
  __builtin_amdgcn_s_wait_asynccnt(0);
#else
  asm volatile("s_wait_asynccnt 0x0" ::: "memory");
#endif
}

// =====================================================================
// Conv1d(k=3, wrap pad) token embedding + positional encoding
// =====================================================================
__global__ void embed_kernel(const float* __restrict__ x, const float* __restrict__ tokw,
                             float* __restrict__ h, bf16_t* __restrict__ hb) {
  long g = (long)blockIdx.x * blockDim.x + threadIdx.x;
  if (g >= (long)B_ * T_ * DM) return;
  int dm = (int)(g % DM);
  long bt = g / DM;
  int t = (int)(bt % T_);
  int b = (int)(bt / T_);
  float acc = 0.f;
  #pragma unroll
  for (int w = 0; w < 3; ++w) {
    int tt = t + w - 1; tt = (tt + T_) & (T_ - 1);
    const float* xr = x + ((long)b * T_ + tt) * F_;
    const float* wr = tokw + (w * F_) * DM + dm;
    #pragma unroll
    for (int f = 0; f < F_; ++f) acc += xr[f] * wr[f * DM];
  }
  int j2 = dm & ~1;
  float div = __expf(-(float)j2 * (9.210340371976184f / 128.f));
  float arg = (float)t * div;
  acc += (dm & 1) ? __cosf(arg) : __sinf(arg);
  h[g] = acc;
  hb[g] = f2bf(acc);
}

// =====================================================================
// mask[b,t] = any(x[b,t,:] != 0)
// =====================================================================
__global__ void mask_kernel(const float* __restrict__ x, float* __restrict__ mask) {
  int g = blockIdx.x * blockDim.x + threadIdx.x;
  if (g >= B_ * T_) return;
  const float* xr = x + (long)g * F_;
  float m = 0.f;
  #pragma unroll
  for (int f = 0; f < F_; ++f) if (xr[f] != 0.f) m = 1.f;
  mask[g] = m;
}

// =====================================================================
// Weight convert+transpose: Wt[n*K+k] = bf16(W[k*N+n])
// =====================================================================
__global__ void convt_kernel(const float* __restrict__ W, bf16_t* __restrict__ Wt,
                             int K, int N) {
  int g = blockIdx.x * blockDim.x + threadIdx.x;
  if (g >= K * N) return;
  int n = g / K, k = g - n * K;
  Wt[g] = f2bf(W[(long)k * N + n]);
}

// =====================================================================
// WMMA GEMM, LDS-staged B: C[M,N] = A x Bt^T + bias
//   block = 8 waves = 128x64 C tile; wave w owns rows w*16..w*16+15, all 64 cols
//   B stripe (64 cols x 32 k, bf16, 4KB) is double-buffered in LDS via
//   global_load_async_to_lds_b128 (each of 256 threads copies 16B),
//   s_wait_asynccnt + barrier, then ds_load_b128 fragment reads.
// Fragment layouts per CDNA5 ISA 7.12.2:
//   A 16x32: lane<16 -> K 0-7,16-23 ; lane>=16 -> K 8-15,24-31 (row = lane%16)
//   B 32x16: lane<16 -> K 0-7,16-23 ; lane>=16 -> K 8-15,24-31      (col = lane%16)
//   C: vgpr r, lanes 0-15 -> M=r, lanes 16-31 -> M=8+r, N = lane%16
// EPI: 0 = f32 out + bias ; 1 = bf16 out + bias + exact GELU
// =====================================================================
template<int EPI>
__global__ void wmma_gemm_kernel(const bf16_t* __restrict__ A,
                                 const bf16_t* __restrict__ Bt,
                                 const float* __restrict__ bias,
                                 float* __restrict__ Cf,
                                 bf16_t* __restrict__ Cb,
                                 int M, int N, int K) {
  __shared__ bf16_t sB[2][BN * 32];   // [buf][col*32 + kk]

  const int tid  = threadIdx.x;
  const int lane = tid & 31;
  const int half = lane >> 4;
  const int sub  = lane & 15;
  const int wv   = tid >> 5;
  const int nbn  = N / BN;
  const int bm   = blockIdx.x / nbn;
  const int bn   = blockIdx.x - bm * nbn;

  // staging: thread tid copies 16B of the B stripe: col = tid/4, k-sub = (tid%4)*8
  const int scol = tid >> 2;
  const int skof = (tid & 3) << 3;
  const bf16_t* sgp = Bt + (long)(bn * BN + scol) * K + skof;
  const int slo = scol * 32 + skof;

  const bf16_t* arow = A + (long)(bm * BM + wv * 16 + sub) * K + half * 8;

  f32x8 acc[4];
  #pragma unroll
  for (int s = 0; s < 4; ++s) acc[s] = (f32x8){0.f,0.f,0.f,0.f,0.f,0.f,0.f,0.f};

  async_copy_b128(sgp, &sB[0][slo]);          // prime buffer 0
  int buf = 0;
  for (int k0 = 0; k0 < K; k0 += 32) {
    wait_async0();
    __syncthreads();                           // stripe visible to all waves
    if (k0 + 32 < K)                           // uniform branch
      async_copy_b128(sgp + k0 + 32, &sB[buf ^ 1][slo]);

    bf16x8 a_lo = *reinterpret_cast<const bf16x8*>(arow + k0);
    bf16x8 a_hi = *reinterpret_cast<const bf16x8*>(arow + k0 + 16);
    bf16x16 af = __builtin_shufflevector(a_lo, a_hi, 0,1,2,3,4,5,6,7,8,9,10,11,12,13,14,15);
    __builtin_prefetch(arow + k0 + 64, 0, 1);  // speculative: safe past end

    #pragma unroll
    for (int s = 0; s < 4; ++s) {
      const bf16_t* bl = &sB[buf][(s * 16 + sub) * 32 + half * 8];
      bf16x8 b_lo = *reinterpret_cast<const bf16x8*>(bl);      // ds_load_b128
      bf16x8 b_hi = *reinterpret_cast<const bf16x8*>(bl + 16);
      bf16x16 bfv = __builtin_shufflevector(b_lo, b_hi, 0,1,2,3,4,5,6,7,8,9,10,11,12,13,14,15);
      acc[s] = __builtin_amdgcn_wmma_f32_16x16x32_bf16(false, af, false, bfv,
                                                       (short)0, acc[s], false, false);
    }
    buf ^= 1;
  }

  const int row = bm * BM + wv * 16 + half * 8;
  #pragma unroll
  for (int s = 0; s < 4; ++s) {
    const int col = bn * BN + s * 16 + sub;
    const float bv = bias ? bias[col] : 0.f;
    #pragma unroll
    for (int r = 0; r < 8; ++r) {
      float v = acc[s][r] + bv;
      if (EPI == 1) v = 0.5f * v * (1.f + erff(v * 0.70710678118f));  // exact GELU
      if (EPI == 0) Cf[(long)(row + r) * N + col] = v;
      else          Cb[(long)(row + r) * N + col] = f2bf(v);
    }
  }
}

// =====================================================================
// Split-K WMMA GEMM with f32 atomic accumulate (classifier head, M=64,N=16)
// =====================================================================
__global__ void wmma_gemm_splitk_kernel(const bf16_t* __restrict__ A,
                                        const bf16_t* __restrict__ Bt,
                                        float* __restrict__ C,
                                        int M, int N, int K, int kchunk) {
  const int lane = threadIdx.x & 31;
  const int half = lane >> 4;
  const int sub  = lane & 15;
  const int wave = blockIdx.x * (blockDim.x >> 5) + (threadIdx.x >> 5);
  const int tilesN = N >> 4;
  const int splits = K / kchunk;
  if (wave >= (M >> 4) * tilesN * splits) return;
  const int sp = wave % splits;
  const int tile = wave / splits;
  const int tm = tile / tilesN;
  const int tn = tile - tm * tilesN;

  const bf16_t* arow = A  + (long)(tm * 16 + sub) * K + half * 8;
  const bf16_t* brow = Bt + (long)(tn * 16 + sub) * K + half * 8;

  f32x8 acc = {0.f, 0.f, 0.f, 0.f, 0.f, 0.f, 0.f, 0.f};
  const int kend = sp * kchunk + kchunk;
  for (int k0 = sp * kchunk; k0 < kend; k0 += 32) {
    bf16x8 a_lo = *reinterpret_cast<const bf16x8*>(arow + k0);
    bf16x8 a_hi = *reinterpret_cast<const bf16x8*>(arow + k0 + 16);
    bf16x8 b_lo = *reinterpret_cast<const bf16x8*>(brow + k0);
    bf16x8 b_hi = *reinterpret_cast<const bf16x8*>(brow + k0 + 16);
    bf16x16 af  = __builtin_shufflevector(a_lo, a_hi, 0,1,2,3,4,5,6,7,8,9,10,11,12,13,14,15);
    bf16x16 bfv = __builtin_shufflevector(b_lo, b_hi, 0,1,2,3,4,5,6,7,8,9,10,11,12,13,14,15);
    __builtin_prefetch(arow + k0 + 64, 0, 1);
    __builtin_prefetch(brow + k0 + 64, 0, 1);
    acc = __builtin_amdgcn_wmma_f32_16x16x32_bf16(false, af, false, bfv,
                                                  (short)0, acc, false, false);
  }
  const int row = tm * 16 + half * 8;
  const int col = tn * 16 + sub;
  #pragma unroll
  for (int r = 0; r < 8; ++r)
    atomicAdd(&C[(long)(row + r) * N + col], acc[r]);
}

// =====================================================================
// LayerNorm (one wave per 128-wide row): h' = LN(h [+ y]) * g + b [* mask]
// =====================================================================
template<int RES, int MASK>
__global__ void ln_kernel(const float* __restrict__ hin, const float* __restrict__ yin,
                          const float* __restrict__ g, const float* __restrict__ bta,
                          const float* __restrict__ mask,
                          float* __restrict__ hout, bf16_t* __restrict__ bout, int rows) {
  int wave = blockIdx.x * (blockDim.x >> 5) + (threadIdx.x >> 5);
  int lane = threadIdx.x & 31;
  if (wave >= rows) return;
  const float* hr = hin + (long)wave * DM;
  float v[4];
  #pragma unroll
  for (int i = 0; i < 4; ++i) {
    int c = lane + 32 * i;
    v[i] = hr[c] + (RES ? yin[(long)wave * DM + c] : 0.f);
  }
  float mean = wave_sum32(v[0] + v[1] + v[2] + v[3]) * (1.f / 128.f);
  float q = 0.f;
  #pragma unroll
  for (int i = 0; i < 4; ++i) { float d = v[i] - mean; q += d * d; }
  q = wave_sum32(q);
  float rs = rsqrtf(q * (1.f / 128.f) + 1e-5f);
  float mk = MASK ? mask[wave] : 1.f;
  #pragma unroll
  for (int i = 0; i < 4; ++i) {
    int c = lane + 32 * i;
    float o = ((v[i] - mean) * rs * g[c] + bta[c]) * mk;
    hout[(long)wave * DM + c] = o;
    bout[(long)wave * DM + c] = f2bf(o);
  }
}

// =====================================================================
// ProbSparse stage 1: M[bh,l] = max_j(q.k_sample) - sum_j/L
// =====================================================================
__global__ void attn_m_kernel(const float* __restrict__ q, const float* __restrict__ k,
                              float* __restrict__ Mbuf) {
  int g = blockIdx.x * blockDim.x + threadIdx.x;
  if (g >= B_ * NH * T_) return;
  int l = g & (T_ - 1);
  int bh = g >> 11;
  int b = bh / NH, h = bh - b * NH;
  const float* qr = q + ((long)b * T_ + l) * DM + h * HD;
  float mx = -3.4e38f, sm = 0.f;
  for (int j = 0; j < UTOP; ++j) {
    unsigned r = 2654435761u * (unsigned)(l * UTOP + j + 1);
    int idx = (int)((r >> 11) & (T_ - 1));
    const float* kr = k + ((long)b * T_ + idx) * DM + h * HD;
    float s = 0.f;
    #pragma unroll
    for (int d = 0; d < HD; ++d) s += qr[d] * kr[d];
    mx = fmaxf(mx, s);
    sm += s;
  }
  Mbuf[(long)bh * T_ + l] = mx - sm * (1.f / (float)T_);
}

// =====================================================================
// top-k (k=40) over M per (b,h): iterative argmax in LDS
// =====================================================================
__global__ void topk_kernel(const float* __restrict__ Mbuf, int* __restrict__ topidx) {
  int bh = blockIdx.x;
  int tid = threadIdx.x;
  __shared__ float sm[T_];
  __shared__ float rmax[256];
  __shared__ int   ridx[256];
  for (int i = tid; i < T_; i += 256) sm[i] = Mbuf[(long)bh * T_ + i];
  __syncthreads();
  for (int it = 0; it < UTOP; ++it) {
    float best = -3.4e38f; int bi = 0;
    for (int i = tid; i < T_; i += 256)
      if (sm[i] > best) { best = sm[i]; bi = i; }
    rmax[tid] = best; ridx[tid] = bi;
    __syncthreads();
    for (int s = 128; s > 0; s >>= 1) {
      if (tid < s && rmax[tid + s] > rmax[tid]) { rmax[tid] = rmax[tid + s]; ridx[tid] = ridx[tid + s]; }
      __syncthreads();
    }
    if (tid == 0) { topidx[bh * UTOP + it] = ridx[0]; sm[ridx[0]] = -3.4e38f; }
    __syncthreads();
  }
}

// =====================================================================
// context init: ctx[b,l,h*32+d] = mean_l V  (bf16, broadcast to all rows)
// =====================================================================
__global__ void vmean_kernel(const float* __restrict__ v, bf16_t* __restrict__ ctx) {
  int bh = blockIdx.x;
  int b = bh / NH, h = bh - b * NH;
  int tid = threadIdx.x;
  int d = tid & 31, c = tid >> 5;
  __shared__ float acc[8][HD];
  float s = 0.f;
  for (int l = c; l < T_; l += 8)
    s += v[((long)b * T_ + l) * DM + h * HD + d];
  acc[c][d] = s;
  __syncthreads();
  if (c == 0) {
    float m = 0.f;
    #pragma unroll
    for (int i = 0; i < 8; ++i) m += acc[i][d];
    acc[0][d] = m * (1.f / (float)T_);
  }
  __syncthreads();
  for (int i = tid; i < T_ * HD; i += 256) {
    int l = i >> 5, dd = i & 31;
    ctx[((long)b * T_ + l) * DM + h * HD + dd] = f2bf(acc[0][dd]);
  }
}

// =====================================================================
// full attention for top-u queries: softmax(q.K^T/sqrt(D)) V
// =====================================================================
__global__ void attn_top_kernel(const float* __restrict__ q, const float* __restrict__ k,
                                const float* __restrict__ v, const int* __restrict__ topidx,
                                bf16_t* __restrict__ ctx) {
  int ui = blockIdx.x;
  int bh = blockIdx.y;
  int b = bh / NH, h = bh - b * NH;
  int tid = threadIdx.x;
  int l = topidx[bh * UTOP + ui];
  __shared__ float qs[HD];
  __shared__ float sc[T_];
  __shared__ float red[256];
  __shared__ float sctx[HD];
  if (tid < HD) { qs[tid] = q[((long)b * T_ + l) * DM + h * HD + tid]; sctx[tid] = 0.f; }
  __syncthreads();
  float lmax = -3.4e38f;
  for (int kk = tid; kk < T_; kk += 256) {
    const float* kr = k + ((long)b * T_ + kk) * DM + h * HD;
    float s = 0.f;
    #pragma unroll
    for (int d = 0; d < HD; ++d) s += qs[d] * kr[d];
    s *= 0.17677669529f;  // 1/sqrt(32)
    sc[kk] = s;
    lmax = fmaxf(lmax, s);
  }
  red[tid] = lmax; __syncthreads();
  for (int s = 128; s > 0; s >>= 1) {
    if (tid < s) red[tid] = fmaxf(red[tid], red[tid + s]);
    __syncthreads();
  }
  float gmax = red[0]; __syncthreads();
  float lsum = 0.f;
  for (int kk = tid; kk < T_; kk += 256) {
    float p = __expf(sc[kk] - gmax);
    sc[kk] = p;
    lsum += p;
  }
  red[tid] = lsum; __syncthreads();
  for (int s = 128; s > 0; s >>= 1) {
    if (tid < s) red[tid] += red[tid + s];
    __syncthreads();
  }
  float gsum = red[0];
  for (int kk = tid; kk < T_; kk += 256) {
    float p = sc[kk];
    const float* vr = v + ((long)b * T_ + kk) * DM + h * HD;
    #pragma unroll
    for (int d = 0; d < HD; ++d) atomicAdd(&sctx[d], p * vr[d]);
  }
  __syncthreads();
  if (tid < HD)
    ctx[((long)b * T_ + l) * DM + h * HD + tid] = f2bf(sctx[tid] / gsum);
}

// =====================================================================
// logits init: out[b,c] = head_b[c]
// =====================================================================
__global__ void init_logits_kernel(float* __restrict__ out, const float* __restrict__ hb) {
  int g = blockIdx.x * blockDim.x + threadIdx.x;
  if (g < B_ * NCLS) out[g] = hb[g & (NCLS - 1)];
}

// =====================================================================
// host orchestration
// =====================================================================
extern "C" void kernel_launch(void* const* d_in, const int* in_sizes, int n_in,
                              void* d_out, int out_size, void* d_ws, size_t ws_size,
                              hipStream_t stream) {
  (void)in_sizes; (void)n_in; (void)out_size;
  const float* x      = (const float*)d_in[0];
  const float* tok_w  = (const float*)d_in[1];
  const float* Wq     = (const float*)d_in[2];
  const float* bq     = (const float*)d_in[3];
  const float* Wk     = (const float*)d_in[4];
  const float* bk     = (const float*)d_in[5];
  const float* Wv     = (const float*)d_in[6];
  const float* bv     = (const float*)d_in[7];
  const float* Wo     = (const float*)d_in[8];
  const float* bo     = (const float*)d_in[9];
  const float* c1_w   = (const float*)d_in[10];
  const float* c1_b   = (const float*)d_in[11];
  const float* c2_w   = (const float*)d_in[12];
  const float* c2_b   = (const float*)d_in[13];
  const float* n1_g   = (const float*)d_in[14];
  const float* n1_b   = (const float*)d_in[15];
  const float* n2_g   = (const float*)d_in[16];
  const float* n2_b   = (const float*)d_in[17];
  const float* norm_g = (const float*)d_in[18];
  const float* norm_b = (const float*)d_in[19];
  const float* head_w = (const float*)d_in[20];
  const float* head_b = (const float*)d_in[21];
  float* logits = (float*)d_out;

  // ---- workspace layout ----
  const size_t ALIGN = 256;
  auto aln = [&](size_t s) { return (s + ALIGN - 1) & ~(ALIGN - 1); };
  const size_t SZ_H   = aln((size_t)B_ * T_ * DM * 4);
  const size_t SZ_HB  = aln((size_t)B_ * T_ * DM * 2);
  const size_t SZ_MID = aln((size_t)B_ * T_ * DFF * 2);
  const size_t SZ_WT  = aln((size_t)T_ * DM * NCLS * 2);
  const size_t SZ_M   = aln((size_t)B_ * NH * T_ * 4);
  const size_t SZ_TI  = aln((size_t)B_ * NH * UTOP * 4);
  const size_t SZ_MK  = aln((size_t)B_ * T_ * 4);
  char* base = (char*)d_ws;
  size_t off = 0;
  float*  h    = (float*)(base + off);  off += SZ_H;
  bf16_t* hb   = (bf16_t*)(base + off); off += SZ_HB;
  float*  qbuf = (float*)(base + off);  off += SZ_H;
  float*  kbuf = (float*)(base + off);  off += SZ_H;
  float*  vbuf = (float*)(base + off);  off += SZ_H;
  float*  ybuf = (float*)(base + off);  off += SZ_H;
  bf16_t* ctxb = (bf16_t*)(base + off); off += SZ_HB;
  bf16_t* midb = (bf16_t*)(base + off); off += SZ_MID;
  bf16_t* wt   = (bf16_t*)(base + off); off += SZ_WT;
  float*  Mbuf = (float*)(base + off);  off += SZ_M;
  int*    tidx = (int*)(base + off);    off += SZ_TI;
  float*  mskb = (float*)(base + off);  off += SZ_MK;
  if (off > ws_size) return;  // workspace too small: refuse deterministically

  const int ROWS = B_ * T_;   // 131072

  auto gemm_f32 = [&](const bf16_t* A, const bf16_t* Bt, const float* bias,
                      float* C, int M, int N, int K) {
    int blocks = (M / BM) * (N / BN);
    wmma_gemm_kernel<0><<<blocks, 256, 0, stream>>>(A, Bt, bias, C, nullptr, M, N, K);
  };
  auto gemm_gelu_bf16 = [&](const bf16_t* A, const bf16_t* Bt, const float* bias,
                            bf16_t* C, int M, int N, int K) {
    int blocks = (M / BM) * (N / BN);
    wmma_gemm_kernel<1><<<blocks, 256, 0, stream>>>(A, Bt, bias, nullptr, C, M, N, K);
  };
  auto convt = [&](const float* W, bf16_t* Wt, int K, int N) {
    int n = K * N;
    convt_kernel<<<(n + 255) / 256, 256, 0, stream>>>(W, Wt, K, N);
  };

  // 1) embedding + PE, input mask
  {
    long n = (long)ROWS * DM;
    embed_kernel<<<(unsigned)((n + 255) / 256), 256, 0, stream>>>(x, tok_w, h, hb);
    mask_kernel<<<(ROWS + 255) / 256, 256, 0, stream>>>(x, mskb);
  }

  // 2) encoder layers
  for (int l = 0; l < ELAY; ++l) {
    const float* Wq_l = Wq + (long)l * DM * DM;
    const float* Wk_l = Wk + (long)l * DM * DM;
    const float* Wv_l = Wv + (long)l * DM * DM;
    const float* Wo_l = Wo + (long)l * DM * DM;

    convt(Wq_l, wt, DM, DM);
    gemm_f32(hb, wt, bq + l * DM, qbuf, ROWS, DM, DM);
    convt(Wk_l, wt, DM, DM);
    gemm_f32(hb, wt, bk + l * DM, kbuf, ROWS, DM, DM);
    convt(Wv_l, wt, DM, DM);
    gemm_f32(hb, wt, bv + l * DM, vbuf, ROWS, DM, DM);

    // ProbSparse attention
    attn_m_kernel<<<(B_ * NH * T_ + 255) / 256, 256, 0, stream>>>(qbuf, kbuf, Mbuf);
    topk_kernel<<<B_ * NH, 256, 0, stream>>>(Mbuf, tidx);
    vmean_kernel<<<B_ * NH, 256, 0, stream>>>(vbuf, ctxb);
    attn_top_kernel<<<dim3(UTOP, B_ * NH), 256, 0, stream>>>(qbuf, kbuf, vbuf, tidx, ctxb);

    // output projection + residual LN1
    convt(Wo_l, wt, DM, DM);
    gemm_f32(ctxb, wt, bo + l * DM, ybuf, ROWS, DM, DM);
    ln_kernel<1, 0><<<(ROWS + 7) / 8, 256, 0, stream>>>(
        h, ybuf, n1_g + l * DM, n1_b + l * DM, nullptr, h, hb, ROWS);

    // FFN + residual LN2
    convt(c1_w + (long)l * DM * DFF, wt, DM, DFF);
    gemm_gelu_bf16(hb, wt, c1_b + l * DFF, midb, ROWS, DFF, DM);
    convt(c2_w + (long)l * DFF * DM, wt, DFF, DM);
    gemm_f32(midb, wt, c2_b + l * DM, ybuf, ROWS, DM, DFF);
    ln_kernel<1, 0><<<(ROWS + 7) / 8, 256, 0, stream>>>(
        h, ybuf, n2_g + l * DM, n2_b + l * DM, nullptr, h, hb, ROWS);
  }

  // 3) final LN with zero-row mask
  ln_kernel<0, 1><<<(ROWS + 7) / 8, 256, 0, stream>>>(
      h, nullptr, norm_g, norm_b, mskb, h, hb, ROWS);

  // 4) classifier head: logits = h.reshape(64, 262144) @ head_w + head_b (split-K)
  convt(head_w, wt, T_ * DM, NCLS);
  init_logits_kernel<<<(B_ * NCLS + 255) / 256, 256, 0, stream>>>(logits, head_b);
  {
    const int K = T_ * DM;          // 262144
    const int KCH = 4096;           // 64 splits -> 256 waves
    int waves = (B_ / 16) * (NCLS / 16) * (K / KCH);
    wmma_gemm_splitk_kernel<<<(waves + 7) / 8, 256, 0, stream>>>(
        hb, wt, logits, B_, NCLS, K, KCH);
  }
}
